// HGTGraphLayer_27960237097203
// MI455X (gfx1250) — compile-verified, matrix operand order checked by reading, not run on hardware
//
#include <hip/hip_runtime.h>
#include <math.h>

// ---------------------------------------------------------------------------
// HGT graph layer for gfx1250 (MI455X).
//  - GEMMs: v_wmma_f32_16x16x32_f16, B operands pre-packed into ISA lane
//    layout (one 32B vector load per fragment), A operands staged to LDS via
//    global_load_async_to_lds_b128 (ASYNCcnt) and read as float4 ds loads.
//  - Edge softmax/aggregation: float4 gathers + f32 atomics (bandwidth-bound).
// ---------------------------------------------------------------------------

#define HID 64
#define NH  4
#define HD  16

typedef __attribute__((ext_vector_type(16))) _Float16 v16h;
typedef __attribute__((ext_vector_type(8)))  float    v8f;

__device__ __forceinline__ v8f wmma_f16(v16h a, v16h b, v8f c) {
  return __builtin_amdgcn_wmma_f32_16x16x32_f16(false, a, false, b, (short)0, c,
                                                false, false);
}

// -------------------- CDNA5 async global->LDS copy -------------------------
// Each lane moves 16B from its global address to its LDS byte offset.
__device__ __forceinline__ void async_copy_b128(unsigned lds_off,
                                                const void* gaddr) {
  asm volatile("global_load_async_to_lds_b128 %0, %1, off"
               :: "v"(lds_off), "v"(gaddr)
               : "memory");
}
__device__ __forceinline__ void wait_async0() {
#if __has_builtin(__builtin_amdgcn_s_wait_asynccnt)
  __builtin_amdgcn_s_wait_asynccnt(0);
#else
  asm volatile("s_wait_asynccnt 0x0" ::: "memory");
#endif
}
// Generic->LDS byte offset (ISA: LDS aperture keeps offset in addr[31:0]).
__device__ __forceinline__ unsigned lds_off_of(const void* p) {
  return (unsigned)(size_t)p;
}
// Copy one 16x64 f32 tile (4KB) global -> LDS, one wave.
__device__ __forceinline__ void async_tile_16x64(float* dstLds,
                                                 const float* src, int lane) {
  const char* g = (const char*)src;
  const unsigned base = lds_off_of(dstLds);
#pragma unroll
  for (int i = 0; i < 8; ++i) {
    const unsigned off = (unsigned)(i * 512 + lane * 16);
    async_copy_b128(base + off, g + off);
  }
  wait_async0();
}

// -------------------- fragment builders ------------------------------------
// A fragment (16x32 f16) from a 16x64 f32 tile in LDS.
// lane(&15)=row M; element j holds K = k0 + (j>>3)*16 + 8*(lane>=16) + (j&7).
__device__ __forceinline__ v16h make_a_frag_sh(const float* t, int k0,
                                               int lane) {
  const int row = lane & 15;
  const int hs  = (lane >> 4) << 3;
  const float* p = t + row * HID + k0 + hs;
  float4 x0 = *(const float4*)(p);
  float4 x1 = *(const float4*)(p + 4);
  float4 x2 = *(const float4*)(p + 16);
  float4 x3 = *(const float4*)(p + 20);
  v16h a;
  a[0]  = (_Float16)x0.x; a[1]  = (_Float16)x0.y;
  a[2]  = (_Float16)x0.z; a[3]  = (_Float16)x0.w;
  a[4]  = (_Float16)x1.x; a[5]  = (_Float16)x1.y;
  a[6]  = (_Float16)x1.z; a[7]  = (_Float16)x1.w;
  a[8]  = (_Float16)x2.x; a[9]  = (_Float16)x2.y;
  a[10] = (_Float16)x2.z; a[11] = (_Float16)x2.w;
  a[12] = (_Float16)x3.x; a[13] = (_Float16)x3.y;
  a[14] = (_Float16)x3.z; a[15] = (_Float16)x3.w;
  return a;
}

// B fragment from pre-packed layout: P[frag][lane][16] halves (32B aligned).
__device__ __forceinline__ v16h load_b_frag(const _Float16* P, int frag,
                                            int lane) {
  return *(const v16h*)(P + ((size_t)frag * 32 + lane) * 16);
}

// Ordered-int trick for float atomic max (addr pre-initialized to -inf).
__device__ __forceinline__ void atomicMaxFloat(float* addr, float val) {
  if (val >= 0.0f)
    atomicMax((int*)addr, __float_as_int(val));
  else
    atomicMin((unsigned int*)addr, (unsigned int)__float_as_int(val));
}

// ---------------------------------------------------------------------------
// K0: pack weights into WMMA B-fragment layout (f16), FiLM gamma/beta.
// P[frag][lane][j]: frag = kc*NT + nt; value = W[kc*32 + (j>>3)*16 +
// 8*(lane>=16) + (j&7)][nt*16 + (lane&15)]    (W is K-major, ldn columns)
// ---------------------------------------------------------------------------
__device__ __forceinline__ void pack_b(const float* W, int ldn, int KK,
                                       _Float16* P, int t, int nthreads) {
  const int NT = ldn >> 4;
  const int KC = KK >> 5;
  const int total = KC * NT * 32 * 16;
  for (int i = t; i < total; i += nthreads) {
    const int j    = i & 15;
    const int lane = (i >> 4) & 31;
    const int frag = i >> 9;
    const int nt   = frag % NT;
    const int kc   = frag / NT;
    const int col  = lane & 15;
    const int hs   = (lane >> 4) << 3;
    const int kk   = kc * 32 + ((j >> 3) << 4) + hs + (j & 7);
    const int nn   = nt * 16 + col;
    P[i] = (_Float16)W[(size_t)kk * ldn + nn];
  }
}

__global__ void hgt_prep_kernel(const float* Wq, const float* Wk,
                                const float* Wv, const float* Wo,
                                const float* W1, const float* W2,
                                const float* task, const float* Wf,
                                const float* bf, _Float16* Pq, _Float16* Pk,
                                _Float16* Pv, _Float16* Po, _Float16* P1,
                                _Float16* P2, float* gmod, float* beta) {
  const int t = threadIdx.x;
  const int nt = blockDim.x;
  pack_b(Wq, HID, HID, Pq, t, nt);
  pack_b(Wk, HID, HID, Pk, t, nt);
  pack_b(Wv, HID, HID, Pv, t, nt);
  pack_b(Wo, HID, HID, Po, t, nt);
  pack_b(W1, 2 * HID, HID, P1, t, nt);
  pack_b(W2, HID, 2 * HID, P2, t, nt);
  if (t < 2 * HID) {
    float acc = bf[t];
#pragma unroll 8
    for (int k = 0; k < HID; ++k) acc += task[k] * Wf[k * 2 * HID + t];
    if (t < HID) gmod[t] = 1.0f + 0.5f * tanhf(acc);
    else         beta[t - HID] = acc;
  }
}

// ---------------------------------------------------------------------------
// K1: init segment buffers (mx=-inf, sums=0, agg=0).
// ---------------------------------------------------------------------------
__global__ void hgt_init_kernel(float* mx, float* sums, float* agg, int n) {
  const int i = blockIdx.x * blockDim.x + threadIdx.x;
  if (i < n * NH) { mx[i] = -INFINITY; sums[i] = 0.0f; }
  if (i < n * HID) agg[i] = 0.0f;
}

// ---------------------------------------------------------------------------
// K2: q,k,v = node @ W{q,k,v} + b  (WMMA, one 16-row tile per wave).
// ---------------------------------------------------------------------------
__global__ __launch_bounds__(128) void hgt_qkv_kernel(
    const float* __restrict__ node,
    const _Float16* __restrict__ Pq, const float* __restrict__ bq,
    const _Float16* __restrict__ Pk, const float* __restrict__ bk,
    const _Float16* __restrict__ Pv, const float* __restrict__ bv,
    float* __restrict__ q, float* __restrict__ k, float* __restrict__ v,
    int n) {
  __shared__ __align__(16) float at[4][16 * HID];
  const int lane = threadIdx.x & 31;
  const int wave = threadIdx.x >> 5;
  const int rowBase = (blockIdx.x * 4 + wave) * 16;
  if (rowBase >= n) return;

  async_tile_16x64(at[wave], node + (size_t)rowBase * HID, lane);
  v16h a0 = make_a_frag_sh(at[wave], 0, lane);
  v16h a1 = make_a_frag_sh(at[wave], 32, lane);

  const _Float16* Ps[3] = {Pq, Pk, Pv};
  const float*    Bs[3] = {bq, bk, bv};
  float*          Os[3] = {q, k, v};

  const int col  = lane & 15;
  const int rofs = rowBase + ((lane >> 4) << 3);

  for (int m = 0; m < 3; ++m) {
#pragma unroll
    for (int nt = 0; nt < 4; ++nt) {
      const float bias = Bs[m][nt * 16 + col];
      v8f c;
#pragma unroll
      for (int r = 0; r < 8; ++r) c[r] = bias;
      c = wmma_f16(a0, load_b_frag(Ps[m], 0 * 4 + nt, lane), c);
      c = wmma_f16(a1, load_b_frag(Ps[m], 1 * 4 + nt, lane), c);
      float* outp = Os[m] + (size_t)rofs * HID + nt * 16 + col;
#pragma unroll
      for (int r = 0; r < 8; ++r) outp[(size_t)r * HID] = c[r];
    }
  }
}

// ---------------------------------------------------------------------------
// K3: per-(edge,head) attention logits + segment max.
// ---------------------------------------------------------------------------
__global__ void hgt_logits_kernel(const float* __restrict__ q,
                                  const float* __restrict__ k,
                                  const int* __restrict__ ei,
                                  const int* __restrict__ etype,
                                  const float* __restrict__ ek_tab,
                                  const float* __restrict__ es_tab,
                                  const float* __restrict__ eg_tab,
                                  float* __restrict__ logits,
                                  float* __restrict__ mx, int ecount) {
  const long long t = (long long)blockIdx.x * blockDim.x + threadIdx.x;
  if (t >= (long long)ecount * NH) return;
  const int e = (int)(t >> 2), h = (int)(t & 3);
  const int s  = ei[e];
  const int d  = ei[ecount + e];
  const int ty = etype[e];

  const float4* qp  = (const float4*)(q + (size_t)d * HID + h * HD);
  const float4* kp  = (const float4*)(k + (size_t)s * HID + h * HD);
  const float4* ekp = (const float4*)(ek_tab + ty * HID + h * HD);
  __builtin_prefetch(qp, 0, 0);   // global_prefetch_b8
  __builtin_prefetch(kp, 0, 0);

  float acc = 0.0f;
#pragma unroll
  for (int i = 0; i < 4; ++i) {
    const float4 qq = qp[i], kk = kp[i], ee = ekp[i];
    acc += qq.x * (kk.x + ee.x) + qq.y * (kk.y + ee.y) +
           qq.z * (kk.z + ee.z) + qq.w * (kk.w + ee.w);
  }

  float g = logf(eg_tab[ty * NH + h]);
  g = fminf(10.0f, fmaxf(-10.0f, g));
  const float lg = acc * 0.25f + es_tab[ty * NH + h] + g;  // 1/sqrt(16)
  logits[t] = lg;
  atomicMaxFloat(&mx[(size_t)d * NH + h], lg);
}

// ---------------------------------------------------------------------------
// K4: ex = exp(logit - mx[dst]); segment sum.
// ---------------------------------------------------------------------------
__global__ void hgt_exp_kernel(const int* __restrict__ ei,
                               const float* __restrict__ mx,
                               float* __restrict__ logits,
                               float* __restrict__ sums, int ecount) {
  const long long t = (long long)blockIdx.x * blockDim.x + threadIdx.x;
  if (t >= (long long)ecount * NH) return;
  const int e = (int)(t >> 2), h = (int)(t & 3);
  const int d = ei[ecount + e];
  float m = mx[(size_t)d * NH + h];
  if (isinf(m)) m = 0.0f;
  const float ex = expf(logits[t] - m);
  logits[t] = ex;
  atomicAdd(&sums[(size_t)d * NH + h], ex);
}

// ---------------------------------------------------------------------------
// K5: messages: agg[dst] += (v[src]+ev) * attn.
// ---------------------------------------------------------------------------
__global__ void hgt_msg_kernel(const int* __restrict__ ei,
                               const int* __restrict__ etype,
                               const float* __restrict__ v,
                               const float* __restrict__ ev_tab,
                               const float* __restrict__ logits,
                               const float* __restrict__ sums,
                               float* __restrict__ agg, int ecount) {
  const long long t = (long long)blockIdx.x * blockDim.x + threadIdx.x;
  if (t >= (long long)ecount * NH) return;
  const int e = (int)(t >> 2), h = (int)(t & 3);
  const int s  = ei[e];
  const int d  = ei[ecount + e];
  const int ty = etype[e];

  const float attn = logits[t] / fmaxf(sums[(size_t)d * NH + h], 1e-12f);
  const float4* vp  = (const float4*)(v + (size_t)s * HID + h * HD);
  const float4* evp = (const float4*)(ev_tab + ty * HID + h * HD);
  float4 vv[4], ee[4];
#pragma unroll
  for (int i = 0; i < 4; ++i) { vv[i] = vp[i]; ee[i] = evp[i]; }
  const float* vf = (const float*)vv;
  const float* ef = (const float*)ee;
  float* ap = agg + (size_t)d * HID + h * HD;
#pragma unroll
  for (int i = 0; i < HD; ++i)
    atomicAdd(&ap[i], (vf[i] + ef[i]) * attn);
}

// ---------------------------------------------------------------------------
// K6: x = LN1(node + FiLM(agg @ Wo + bo)).
// ---------------------------------------------------------------------------
__global__ __launch_bounds__(128) void hgt_attnout_kernel(
    const float* __restrict__ agg, const _Float16* __restrict__ Po,
    const float* __restrict__ bo, const float* __restrict__ gmod,
    const float* __restrict__ beta, const float* __restrict__ node,
    const float* __restrict__ ln1g, const float* __restrict__ ln1b,
    float* __restrict__ x, int n) {
  __shared__ __align__(16) float at[4][16 * HID];   // agg tile (A operand)
  __shared__ __align__(16) float rt[4][16 * HID];   // node tile / LN staging
  __shared__ float mu[4][16], rs[4][16];

  const int lane = threadIdx.x & 31;
  const int wave = threadIdx.x >> 5;
  const int rowBase = (blockIdx.x * 4 + wave) * 16;
  if (rowBase >= n) return;

  async_tile_16x64(at[wave], agg + (size_t)rowBase * HID, lane);
  async_tile_16x64(rt[wave], node + (size_t)rowBase * HID, lane);

  v16h a0 = make_a_frag_sh(at[wave], 0, lane);
  v16h a1 = make_a_frag_sh(at[wave], 32, lane);

  const int col = lane & 15;
  const int rhi = (lane >> 4) << 3;

#pragma unroll
  for (int nt = 0; nt < 4; ++nt) {
    const int n0 = nt * 16;
    const float bias = bo[n0 + col];
    v8f c;
#pragma unroll
    for (int r = 0; r < 8; ++r) c[r] = bias;
    c = wmma_f16(a0, load_b_frag(Po, 0 * 4 + nt, lane), c);
    c = wmma_f16(a1, load_b_frag(Po, 1 * 4 + nt, lane), c);
    const float gm = gmod[n0 + col], bt = beta[n0 + col];
#pragma unroll
    for (int r = 0; r < 8; ++r) {
      const int idx = (r + rhi) * HID + n0 + col;
      rt[wave][idx] = c[r] * gm + bt + rt[wave][idx];   // + node residual
    }
  }

  // per-row layer-norm (wave-local; LDS ops in-order per wave)
  if (lane < 16) {
    const float* row = &rt[wave][lane * HID];
    float m = 0.0f;
#pragma unroll
    for (int j = 0; j < HID; ++j) m += row[j];
    m *= (1.0f / HID);
    float var = 0.0f;
#pragma unroll
    for (int j = 0; j < HID; ++j) { const float dd = row[j] - m; var += dd * dd; }
    var *= (1.0f / HID);
    mu[wave][lane] = m;
    rs[wave][lane] = rsqrtf(var + 1e-5f);
  }
#pragma unroll
  for (int idx = lane; idx < 16 * HID; idx += 32) {
    const int row = idx >> 6, cc = idx & 63;
    x[(size_t)(rowBase + row) * HID + cc] =
        (rt[wave][idx] - mu[wave][row]) * rs[wave][row] * ln1g[cc] + ln1b[cc];
  }
}

// ---------------------------------------------------------------------------
// K7: out = LN2(x + gelu(x@W1+b1)@W2+b2); hidden kept in LDS in A-fragment
// layout so second-GEMM A operands are contiguous 32B ds loads.
// ---------------------------------------------------------------------------
__global__ __launch_bounds__(128) void hgt_ffn_kernel(
    const float* __restrict__ x, const _Float16* __restrict__ P1,
    const float* __restrict__ b1, const _Float16* __restrict__ P2,
    const float* __restrict__ b2, const float* __restrict__ ln2g,
    const float* __restrict__ ln2b, float* __restrict__ out, int n) {
  __shared__ __align__(16) float    xt[4][16 * HID];         // x tile
  __shared__ __align__(32) _Float16 hfrag[4][4 * 32 * 16];   // h in frag layout
  __shared__ float mu[4][16], rs[4][16];

  const int lane = threadIdx.x & 31;
  const int wave = threadIdx.x >> 5;
  const int rowBase = (blockIdx.x * 4 + wave) * 16;
  if (rowBase >= n) return;

  async_tile_16x64(xt[wave], x + (size_t)rowBase * HID, lane);
  v16h a0 = make_a_frag_sh(xt[wave], 0, lane);
  v16h a1 = make_a_frag_sh(xt[wave], 32, lane);

  const int col = lane & 15;
  const int rhi = (lane >> 4) << 3;

  // h = gelu(x @ W1 + b1): 8 col tiles; scatter into A-fragment layout.
#pragma unroll
  for (int nt = 0; nt < 8; ++nt) {
    const float bias = b1[nt * 16 + col];
    v8f c;
#pragma unroll
    for (int r = 0; r < 8; ++r) c[r] = bias;
    c = wmma_f16(a0, load_b_frag(P1, 0 * 8 + nt, lane), c);
    c = wmma_f16(a1, load_b_frag(P1, 1 * 8 + nt, lane), c);
#pragma unroll
    for (int r = 0; r < 8; ++r) {
      const int row  = r + rhi;
      const int kcol = nt * 16 + col;
      const int kc   = kcol >> 5;
      const int kk   = kcol & 31;
      const int j    = ((kk >> 4) << 3) | (kk & 7);
      const int rl   = row + (((kk >> 3) & 1) << 4);
      const float g  = c[r];
      hfrag[wave][(kc * 32 + rl) * 16 + j] =
          (_Float16)(0.5f * g * (1.0f + erff(g * 0.70710678118654752f)));
    }
  }

  // y = h @ W2 + b2 + x; accumulate into xt in place.
#pragma unroll
  for (int nt = 0; nt < 4; ++nt) {
    const float bias = b2[nt * 16 + col];
    v8f c;
#pragma unroll
    for (int r = 0; r < 8; ++r) c[r] = bias;
#pragma unroll
    for (int kc = 0; kc < 4; ++kc) {
      v16h af = *(const v16h*)(&hfrag[wave][(kc * 32 + lane) * 16]);
      c = wmma_f16(af, load_b_frag(P2, kc * 4 + nt, lane), c);
    }
#pragma unroll
    for (int r = 0; r < 8; ++r) {
      const int idx = (r + rhi) * HID + nt * 16 + col;
      xt[wave][idx] = c[r] + xt[wave][idx];
    }
  }

  if (lane < 16) {
    const float* row = &xt[wave][lane * HID];
    float m = 0.0f;
#pragma unroll
    for (int j = 0; j < HID; ++j) m += row[j];
    m *= (1.0f / HID);
    float var = 0.0f;
#pragma unroll
    for (int j = 0; j < HID; ++j) { const float dd = row[j] - m; var += dd * dd; }
    var *= (1.0f / HID);
    mu[wave][lane] = m;
    rs[wave][lane] = rsqrtf(var + 1e-5f);
  }
#pragma unroll
  for (int idx = lane; idx < 16 * HID; idx += 32) {
    const int row = idx >> 6, cc = idx & 63;
    out[(size_t)(rowBase + row) * HID + cc] =
        (xt[wave][idx] - mu[wave][row]) * rs[wave][row] * ln2g[cc] + ln2b[cc];
  }
}

// ---------------------------------------------------------------------------
// Launch
// ---------------------------------------------------------------------------
extern "C" void kernel_launch(void* const* d_in, const int* in_sizes, int n_in,
                              void* d_out, int out_size, void* d_ws,
                              size_t ws_size, hipStream_t stream) {
  const float* node  = (const float*)d_in[0];
  const int*   ei    = (const int*)d_in[1];
  const int*   etype = (const int*)d_in[2];
  const float* task  = (const float*)d_in[3];
  const float* Wq = (const float*)d_in[4];  const float* bq = (const float*)d_in[5];
  const float* Wk = (const float*)d_in[6];  const float* bk = (const float*)d_in[7];
  const float* Wv = (const float*)d_in[8];  const float* bv = (const float*)d_in[9];
  const float* ek_tab = (const float*)d_in[10];
  const float* ev_tab = (const float*)d_in[11];
  const float* es_tab = (const float*)d_in[12];
  const float* eg_tab = (const float*)d_in[13];
  const float* Wo = (const float*)d_in[14]; const float* bo = (const float*)d_in[15];
  const float* ln1g = (const float*)d_in[16]; const float* ln1b = (const float*)d_in[17];
  const float* W1 = (const float*)d_in[18]; const float* b1 = (const float*)d_in[19];
  const float* W2 = (const float*)d_in[20]; const float* b2 = (const float*)d_in[21];
  const float* ln2g = (const float*)d_in[22]; const float* ln2b = (const float*)d_in[23];
  const float* Wf = (const float*)d_in[24]; const float* bf = (const float*)d_in[25];

  const int N = in_sizes[0] / HID;
  const int E = in_sizes[2];

  char* p = (char*)d_ws;
  auto carve = [&](size_t bytes) {
    void* r = (void*)p;
    p += (bytes + 255) & ~(size_t)255;
    return r;
  };
  float* q      = (float*)carve((size_t)N * HID * 4);
  float* k      = (float*)carve((size_t)N * HID * 4);
  float* v      = (float*)carve((size_t)N * HID * 4);
  float* agg    = (float*)carve((size_t)N * HID * 4);
  float* x      = (float*)carve((size_t)N * HID * 4);
  float* logits = (float*)carve((size_t)E * NH * 4);
  float* mx     = (float*)carve((size_t)N * NH * 4);
  float* sums   = (float*)carve((size_t)N * NH * 4);
  _Float16* Pq = (_Float16*)carve(2 * 4 * 32 * 16 * 2);   // 8 frags
  _Float16* Pk = (_Float16*)carve(2 * 4 * 32 * 16 * 2);
  _Float16* Pv = (_Float16*)carve(2 * 4 * 32 * 16 * 2);
  _Float16* Po = (_Float16*)carve(2 * 4 * 32 * 16 * 2);
  _Float16* P1 = (_Float16*)carve(2 * 8 * 32 * 16 * 2);   // 16 frags
  _Float16* P2 = (_Float16*)carve(4 * 4 * 32 * 16 * 2);   // 16 frags
  float* gmod = (float*)carve(HID * 4);
  float* beta = (float*)carve(HID * 4);

  hgt_prep_kernel<<<1, 256, 0, stream>>>(Wq, Wk, Wv, Wo, W1, W2, task, Wf, bf,
                                         Pq, Pk, Pv, Po, P1, P2, gmod, beta);

  {
    const int total = N * HID;
    hgt_init_kernel<<<(total + 255) / 256, 256, 0, stream>>>(mx, sums, agg, N);
  }

  const int tiles = (N + 15) / 16;
  const int gemmBlocks = (tiles + 3) / 4;
  hgt_qkv_kernel<<<gemmBlocks, 128, 0, stream>>>(node, Pq, bq, Pk, bk, Pv, bv,
                                                 q, k, v, N);

  {
    const long long eh = (long long)E * NH;
    const int blocks = (int)((eh + 255) / 256);
    hgt_logits_kernel<<<blocks, 256, 0, stream>>>(q, k, ei, etype, ek_tab,
                                                  es_tab, eg_tab, logits, mx, E);
    hgt_exp_kernel<<<blocks, 256, 0, stream>>>(ei, mx, logits, sums, E);
    hgt_msg_kernel<<<blocks, 256, 0, stream>>>(ei, etype, v, ev_tab, logits,
                                               sums, agg, E);
  }

  hgt_attnout_kernel<<<gemmBlocks, 128, 0, stream>>>(agg, Po, bo, gmod, beta,
                                                     node, ln1g, ln1b, x, N);
  hgt_ffn_kernel<<<gemmBlocks, 128, 0, stream>>>(x, P1, b1, P2, b2, ln2g, ln2b,
                                                 (float*)d_out, N);
}
